// ConvolutionGGN_25194278158759
// MI455X (gfx1250) — matrix operand order repaired and checked by reading, not compiled, not run on hardware
//
#include <hip/hip_runtime.h>
#include <hip/hip_bf16.h>
#include <cstdint>
#include <cstddef>

// ---------------------------------------------------------------------------
// Types for CDNA5 WMMA (wave32): D(f32 16x16) = A(bf16 16x32) x B(bf16 32x16) + C
// ---------------------------------------------------------------------------
typedef __attribute__((ext_vector_type(16))) __bf16 bf16x16;
typedef __attribute__((ext_vector_type(8)))  float  f32x8;

__device__ __forceinline__ bf16x16 load_frag_a(const __bf16* p)
{
    // A 16x32 bf16 fragment (per ISA 7.12.2): elems 0..7 at K+0..7 (lane-half
    // offset applied by caller), elems 8..15 at K+16..23.
    union { uint4 q[2]; bf16x16 v; } f;
    f.q[0] = *(const uint4*)(p);
    f.q[1] = *(const uint4*)(p + 16);
    return f.v;
}

__device__ __forceinline__ bf16x16 load_frag_b(const __bf16* p)
{
    // B 32x16 bf16 fragment: elems 0..7 at K+0..7, elems 8..15 at K+16..23 (half offset by caller).
    union { uint4 q[2]; bf16x16 v; } f;
    f.q[0] = *(const uint4*)(p);
    f.q[1] = *(const uint4*)(p + 16);
    return f.v;
}

// ---------------------------------------------------------------------------
// GEMM: C[M,N] = A[M,K](bf16 row-major) * Bt[N,K](bf16, pre-transposed B)
//       + bias[N], optional ReLU.
// Wave computes a 32x(16*NT) tile: 2 A-row tiles x NT B-subtiles = 2*NT WMMAs
// per 32-deep K step, with register double-buffering so fragment loads for
// k+32 are in flight while WMMAs for k execute. All branches are scalar-
// uniform so EXEC is all-ones at every WMMA (ISA requirement).
// Requires: M%32==0, K%32==0, N%(16*NT)==0.
// ---------------------------------------------------------------------------
template <int NT>
__global__ __launch_bounds__(256)
void gemm_bf16_wmma(const __bf16* __restrict__ A,
                    const __bf16* __restrict__ Bt,
                    const float*  __restrict__ bias,
                    float*        __restrict__ C,
                    int M, int N, int K, int relu)
{
    const int wave   = blockIdx.x * (blockDim.x >> 5) + (threadIdx.x >> 5);
    const int lane   = threadIdx.x & 31;
    const int mtiles = M >> 5;
    const int ntiles = N / (16 * NT);
    if (wave >= mtiles * ntiles) return;

    const int mt  = wave % mtiles;
    const int nt  = wave / mtiles;
    const int hi  = lane >> 4;      // lane half (0/1)
    const int l15 = lane & 15;
    const int n0  = nt * (16 * NT);

    const __bf16* __restrict__ arow0 = A + (size_t)(mt * 32 + l15) * K + hi * 8;
    const __bf16* __restrict__ arow1 = arow0 + (size_t)16 * K;
    const __bf16* brow[NT];
#pragma unroll
    for (int j = 0; j < NT; ++j)
        brow[j] = Bt + (size_t)(n0 + j * 16 + l15) * K + hi * 8;

    f32x8   acc[2][NT] = {};
    bf16x16 fa[2][2];
    bf16x16 fb[2][NT];

    auto loadbuf = [&](int buf, int koff) {
        fa[buf][0] = load_frag_a(arow0 + koff);
        fa[buf][1] = load_frag_a(arow1 + koff);
#pragma unroll
        for (int j = 0; j < NT; ++j) fb[buf][j] = load_frag_b(brow[j] + koff);
    };
    auto mmabuf = [&](int buf) {
#pragma unroll
        for (int j = 0; j < NT; ++j) {
            acc[0][j] = __builtin_amdgcn_wmma_f32_16x16x32_bf16(
                false, fa[buf][0], false, fb[buf][j], (short)0, acc[0][j], false, false);
            acc[1][j] = __builtin_amdgcn_wmma_f32_16x16x32_bf16(
                false, fa[buf][1], false, fb[buf][j], (short)0, acc[1][j], false, false);
        }
    };

    loadbuf(0, 0);
    for (int k0 = 0; k0 < K; k0 += 64) {
        if (k0 + 32 < K) loadbuf(1, k0 + 32);   // prefetch while buf0 computes
        mmabuf(0);
        if (k0 + 64 < K) loadbuf(0, k0 + 64);   // prefetch while buf1 computes
        if (k0 + 32 < K) mmabuf(1);
    }

#pragma unroll
    for (int m = 0; m < 2; ++m) {
#pragma unroll
        for (int j = 0; j < NT; ++j) {
            const int col = n0 + j * 16 + l15;
            const float bv = bias ? bias[col] : 0.0f;
#pragma unroll
            for (int r = 0; r < 8; ++r) {
                const int orow = mt * 32 + m * 16 + r + hi * 8;
                float v = acc[m][j][r] + bv;
                if (relu) v = v > 0.0f ? v : 0.0f;
                C[(size_t)orow * N + col] = v;
            }
        }
    }
}

// ---------------------------------------------------------------------------
// Elementwise / data-movement kernels
// ---------------------------------------------------------------------------
__global__ __launch_bounds__(256)
void cvt_f32_bf16(const float* __restrict__ in, __bf16* __restrict__ out, size_t n)
{
    size_t i = (size_t)blockIdx.x * blockDim.x + threadIdx.x;
    if (i < n) out[i] = (__bf16)in[i];
}

// W[K,N] fp32 -> Wt[N,K] bf16 (k fastest so writes are coalesced)
__global__ __launch_bounds__(256)
void transpose_cvt_bf16(const float* __restrict__ W, __bf16* __restrict__ Wt,
                        int K, int N)
{
    size_t i = (size_t)blockIdx.x * blockDim.x + threadIdx.x;
    size_t total = (size_t)K * N;
    if (i >= total) return;
    int k = (int)(i % K);
    int n = (int)(i / K);
    Wt[i] = (__bf16)W[(size_t)k * N + n];
}

// layer-0 edge embedding: E has 1 feature -> outer product with w[1,F] + b[F]
__global__ __launch_bounds__(256)
void emb_e0_kernel(const float* __restrict__ E, const float* __restrict__ w,
                   const float* __restrict__ b, float* __restrict__ out,
                   size_t total, int F)
{
    size_t i = (size_t)blockIdx.x * blockDim.x + threadIdx.x;
    if (i >= total) return;
    size_t e = i / F;
    int    f = (int)(i % F);
    out[i] = E[e] * w[f] + b[f];
}

// e = CE + DX[src] + EX[dst]  (in place over CE buffer)
__global__ __launch_bounds__(256)
void gate_add_kernel(float* __restrict__ eg, const float* __restrict__ DX,
                     const float* __restrict__ EX, const int* __restrict__ src,
                     const int* __restrict__ dst, size_t total, int F)
{
    size_t i = (size_t)blockIdx.x * blockDim.x + threadIdx.x;
    if (i >= total) return;
    size_t k = i / F;
    int    f = (int)(i % F);
    eg[i] += DX[(size_t)src[k] * F + f] + EX[(size_t)dst[k] * F + f];
}

// agg = segment_sum(e * BX[src], dst) exploiting the complete-graph edge
// ordering (s-major, d skips s):  edge(g,s,d) = g*132 + s*11 + (d - (d>s)).
// Then h = relu(AX + agg), written in place over AX. Fully deterministic.
__global__ __launch_bounds__(256)
void agg_h_kernel(float* __restrict__ AX, const float* __restrict__ eg,
                  const float* __restrict__ BX, size_t total, int F)
{
    size_t i = (size_t)blockIdx.x * blockDim.x + threadIdx.x;
    if (i >= total) return;
    int n  = (int)(i / F);
    int f  = (int)(i % F);
    int g  = n / 12;
    int dn = n % 12;
    size_t gbase = (size_t)g * 132;
    float acc = 0.0f;
#pragma unroll
    for (int s = 0; s < 12; ++s) {
        if (s == dn) continue;
        size_t eidx = gbase + (size_t)s * 11 + (size_t)(dn - (dn > s ? 1 : 0));
        acc += eg[eidx * F + f] * BX[((size_t)g * 12 + s) * F + f];
    }
    float h = AX[i] + acc;
    AX[i] = h > 0.0f ? h : 0.0f;
}

// training-mode BatchNorm stats: one block per feature, biased variance.
__global__ __launch_bounds__(256)
void bn_stats_kernel(const float* __restrict__ x, int rows, int F,
                     float* __restrict__ mu, float* __restrict__ rstd)
{
    int f = blockIdx.x;
    float s = 0.0f, ss = 0.0f;
    for (int r = threadIdx.x; r < rows; r += blockDim.x) {
        float v = x[(size_t)r * F + f];
        s += v; ss += v * v;
    }
    __shared__ float sh0[256];
    __shared__ float sh1[256];
    sh0[threadIdx.x] = s; sh1[threadIdx.x] = ss;
    __syncthreads();
    for (int o = blockDim.x >> 1; o > 0; o >>= 1) {
        if ((int)threadIdx.x < o) {
            sh0[threadIdx.x] += sh0[threadIdx.x + o];
            sh1[threadIdx.x] += sh1[threadIdx.x + o];
        }
        __syncthreads();
    }
    if (threadIdx.x == 0) {
        float m = sh0[0] / rows;
        float v = sh1[0] / rows - m * m;
        v = v < 0.0f ? 0.0f : v;
        mu[f]   = m;
        rstd[f] = rsqrtf(v + 1e-5f);
    }
}

// out = relu( xres + (h - mu)*rstd*g + b )
__global__ __launch_bounds__(256)
void bn_apply_kernel(const float* __restrict__ xres, const float* __restrict__ h,
                     const float* __restrict__ mu, const float* __restrict__ rstd,
                     const float* __restrict__ g, const float* __restrict__ b,
                     float* __restrict__ out, size_t total, int F)
{
    size_t i = (size_t)blockIdx.x * blockDim.x + threadIdx.x;
    if (i >= total) return;
    int f = (int)(i % F);
    float v = xres[i] + (h[i] - mu[f]) * rstd[f] * g[f] + b[f];
    out[i] = v > 0.0f ? v : 0.0f;
}

// parents = mean over nodes {0,2,4,6,8,10}; children = mean over {1,3,5,7,9}
__global__ __launch_bounds__(256)
void readout_kernel(const float* __restrict__ H, float* __restrict__ parent,
                    float* __restrict__ child, int F, int colOff, int FT, int B)
{
    size_t i = (size_t)blockIdx.x * blockDim.x + threadIdx.x;
    if (i >= (size_t)B * F) return;
    int b = (int)(i / F);
    int f = (int)(i % F);
    const float* base = H + ((size_t)b * 12) * F + f;
    float ps = 0.0f, cs = 0.0f;
#pragma unroll
    for (int p = 0; p < 11; p += 2) ps += base[(size_t)p * F];
#pragma unroll
    for (int p = 1; p < 10; p += 2) cs += base[(size_t)p * F];
    parent[(size_t)b * FT + colOff + f] = ps * (1.0f / 6.0f);
    child [(size_t)b * FT + colOff + f] = cs * (1.0f / 5.0f);
}

// concat [parent | child] -> bf16  ([B, 2*FT])
__global__ __launch_bounds__(256)
void cat_cvt_kernel(const float* __restrict__ parent, const float* __restrict__ child,
                    __bf16* __restrict__ out, int B, int FT)
{
    size_t i = (size_t)blockIdx.x * blockDim.x + threadIdx.x;
    if (i >= (size_t)B * 2 * FT) return;
    int b = (int)(i / (2 * FT));
    int c = (int)(i % (2 * FT));
    float v = (c < FT) ? parent[(size_t)b * FT + c]
                       : child [(size_t)b * FT + (c - FT)];
    out[i] = (__bf16)v;
}

// y[b, 0..1] = center[b,:] @ w2[:,2] + b2   (K=2720, tiny -> block reduction)
__global__ __launch_bounds__(256)
void final_y_kernel(const float* __restrict__ center, const float* __restrict__ w2,
                    const float* __restrict__ b2, float* __restrict__ y, int K)
{
    int b = blockIdx.x;
    float a0 = 0.0f, a1 = 0.0f;
    for (int k = threadIdx.x; k < K; k += blockDim.x) {
        float c = center[(size_t)b * K + k];
        a0 += c * w2[(size_t)k * 2 + 0];
        a1 += c * w2[(size_t)k * 2 + 1];
    }
    __shared__ float s0[256];
    __shared__ float s1[256];
    s0[threadIdx.x] = a0; s1[threadIdx.x] = a1;
    __syncthreads();
    for (int o = blockDim.x >> 1; o > 0; o >>= 1) {
        if ((int)threadIdx.x < o) {
            s0[threadIdx.x] += s0[threadIdx.x + o];
            s1[threadIdx.x] += s1[threadIdx.x + o];
        }
        __syncthreads();
    }
    if (threadIdx.x == 0) {
        y[(size_t)b * 2 + 0] = s0[0] + b2[0];
        y[(size_t)b * 2 + 1] = s1[0] + b2[1];
    }
}

// ---------------------------------------------------------------------------
// Host-side orchestration
// ---------------------------------------------------------------------------
static inline dim3 gsz(size_t n) { return dim3((unsigned)((n + 255) / 256)); }

static void launch_gemm(const __bf16* A, const __bf16* Bt, const float* bias,
                        float* C, int M, int N, int K, int relu, hipStream_t s)
{
    if (N % 64 == 0) {
        int waves = (M / 32) * (N / 64);
        gemm_bf16_wmma<4><<<(waves + 7) / 8, 256, 0, s>>>(A, Bt, bias, C, M, N, K, relu);
    } else if (N % 32 == 0) {
        int waves = (M / 32) * (N / 32);
        gemm_bf16_wmma<2><<<(waves + 7) / 8, 256, 0, s>>>(A, Bt, bias, C, M, N, K, relu);
    } else {
        int waves = (M / 32) * (N / 16);
        gemm_bf16_wmma<1><<<(waves + 7) / 8, 256, 0, s>>>(A, Bt, bias, C, M, N, K, relu);
    }
}

extern "C" void kernel_launch(void* const* d_in, const int* in_sizes, int n_in,
                              void* d_out, int out_size, void* d_ws, size_t ws_size,
                              hipStream_t stream)
{
    (void)in_sizes; (void)n_in; (void)out_size; (void)ws_size;

    const int NN = 3072;     // nodes
    const int NE = 33792;    // edges
    const int B  = 256;      // graphs
    const int FT = 2720;     // concat feature total
    const int H_DIMS[5] = {2048, 2048, 512, 128, 32};

    const float* X   = (const float*)d_in[0];
    const float* Ein = (const float*)d_in[1];
    const int*   src = (const int*)d_in[2];
    const int*   dst = (const int*)d_in[3];
    // emb_h[i]: w=d_in[4+2i], b=d_in[5+2i]; emb_e[i]: w=d_in[12+2i], b=d_in[13+2i]
    // conv[i] base = 20 + 14*i: Aw,Ab,Bw,Bb,Cw,Cb,Dw,Db,Ew,Eb,bnh_g,bnh_b,bne_g,bne_b
    // mlp: w1=d_in[76], b1=d_in[77], w2=d_in[78], b2=d_in[79]

    // ---- workspace carve ----
    size_t off = 0;
    auto carve = [&](size_t bytes) -> void* {
        void* p = (char*)d_ws + off;
        off += (bytes + 255) & ~(size_t)255;
        return p;
    };
    float*  eEmb  = (float*) carve((size_t)NE * 2048 * 4);
    float*  eGate = (float*) carve((size_t)NE * 2048 * 4);
    float*  eCur  = (float*) carve((size_t)NE * 2048 * 4);
    __bf16* eBf   = (__bf16*)carve((size_t)NE * 2048 * 2);
    float*  hEmb  = (float*) carve((size_t)NN * 2048 * 4);
    float*  AX    = (float*) carve((size_t)NN * 2048 * 4);
    float*  BXb   = (float*) carve((size_t)NN * 2048 * 4);
    float*  DX    = (float*) carve((size_t)NN * 2048 * 4);
    float*  EX    = (float*) carve((size_t)NN * 2048 * 4);
    float*  hCurW = (float*) carve((size_t)NN * 2048 * 4);
    __bf16* hBf   = (__bf16*)carve((size_t)NN * 2048 * 2);
    __bf16* Wt    = (__bf16*)carve((size_t)2720 * 5440 * 2); // covers 2048x2048 too
    float*  muh   = (float*) carve(2048 * 4);
    float*  rsh   = (float*) carve(2048 * 4);
    float*  mue   = (float*) carve(2048 * 4);
    float*  rse   = (float*) carve(2048 * 4);
    __bf16* catBf = (__bf16*)carve((size_t)B * 5440 * 2);

    // ---- d_out layout: y[512] | parent[B*FT] | child[B*FT] | center[B*FT] ----
    float* dy      = (float*)d_out;
    float* dparent = dy + 512;
    float* dchild  = dparent + (size_t)B * FT;
    float* dcenter = dchild  + (size_t)B * FT;

    const float* Hcur = X;
    int din = 2048;
    int colOff = 0;

    for (int i = 0; i < 4; ++i) {
        const int dout = H_DIMS[i + 1];
        const float* wh = (const float*)d_in[4 + 2 * i];
        const float* bh = (const float*)d_in[5 + 2 * i];
        const float* we = (const float*)d_in[12 + 2 * i];
        const float* be = (const float*)d_in[13 + 2 * i];
        const int cb = 20 + 14 * i;
        const float* Aw = (const float*)d_in[cb + 0];
        const float* Ab = (const float*)d_in[cb + 1];
        const float* Bw = (const float*)d_in[cb + 2];
        const float* Bb = (const float*)d_in[cb + 3];
        const float* Cw = (const float*)d_in[cb + 4];
        const float* Cb = (const float*)d_in[cb + 5];
        const float* Dw = (const float*)d_in[cb + 6];
        const float* Db = (const float*)d_in[cb + 7];
        const float* Ew = (const float*)d_in[cb + 8];
        const float* Eb = (const float*)d_in[cb + 9];
        const float* bnhg = (const float*)d_in[cb + 10];
        const float* bnhb = (const float*)d_in[cb + 11];
        const float* bneg = (const float*)d_in[cb + 12];
        const float* bneb = (const float*)d_in[cb + 13];

        // 1) node embedding: hEmb = bf16(Hcur) @ wh + bh
        size_t nh = (size_t)NN * din;
        cvt_f32_bf16<<<gsz(nh), 256, 0, stream>>>(Hcur, hBf, nh);
        transpose_cvt_bf16<<<gsz((size_t)din * dout), 256, 0, stream>>>(wh, Wt, din, dout);
        launch_gemm(hBf, Wt, bh, hEmb, NN, dout, din, 0, stream);

        // 2) edge embedding
        if (i == 0) {
            size_t te = (size_t)NE * dout;
            emb_e0_kernel<<<gsz(te), 256, 0, stream>>>(Ein, we, be, eEmb, te, dout);
        } else {
            size_t ne = (size_t)NE * din;
            cvt_f32_bf16<<<gsz(ne), 256, 0, stream>>>(eCur, eBf, ne);
            transpose_cvt_bf16<<<gsz((size_t)din * dout), 256, 0, stream>>>(we, Wt, din, dout);
            launch_gemm(eBf, Wt, be, eEmb, NE, dout, din, 0, stream);
        }

        // 3) node-side conv GEMMs: AX, BX, DX, EX = hEmb @ {A,B,D,E}w + bias
        size_t nh2 = (size_t)NN * dout;
        cvt_f32_bf16<<<gsz(nh2), 256, 0, stream>>>(hEmb, hBf, nh2);
        transpose_cvt_bf16<<<gsz((size_t)dout * dout), 256, 0, stream>>>(Aw, Wt, dout, dout);
        launch_gemm(hBf, Wt, Ab, AX,  NN, dout, dout, 0, stream);
        transpose_cvt_bf16<<<gsz((size_t)dout * dout), 256, 0, stream>>>(Bw, Wt, dout, dout);
        launch_gemm(hBf, Wt, Bb, BXb, NN, dout, dout, 0, stream);
        transpose_cvt_bf16<<<gsz((size_t)dout * dout), 256, 0, stream>>>(Dw, Wt, dout, dout);
        launch_gemm(hBf, Wt, Db, DX,  NN, dout, dout, 0, stream);
        transpose_cvt_bf16<<<gsz((size_t)dout * dout), 256, 0, stream>>>(Ew, Wt, dout, dout);
        launch_gemm(hBf, Wt, Eb, EX,  NN, dout, dout, 0, stream);

        // 4) edge-side conv GEMM: eGate = bf16(eEmb) @ Cw + Cb
        size_t ne2 = (size_t)NE * dout;
        cvt_f32_bf16<<<gsz(ne2), 256, 0, stream>>>(eEmb, eBf, ne2);
        transpose_cvt_bf16<<<gsz((size_t)dout * dout), 256, 0, stream>>>(Cw, Wt, dout, dout);
        launch_gemm(eBf, Wt, Cb, eGate, NE, dout, dout, 0, stream);

        // 5) gate: e = CE + DX[src] + EX[dst]   (in place)
        gate_add_kernel<<<gsz(ne2), 256, 0, stream>>>(eGate, DX, EX, src, dst, ne2, dout);

        // 6) agg + h = relu(AX + segment_sum(e*BX[src], dst))  (in place over AX)
        agg_h_kernel<<<gsz(nh2), 256, 0, stream>>>(AX, eGate, BXb, nh2, dout);

        // 7) BatchNorm stats + apply (residual on embedded features) + relu
        bn_stats_kernel<<<dout, 256, 0, stream>>>(AX,    NN, dout, muh, rsh);
        bn_stats_kernel<<<dout, 256, 0, stream>>>(eGate, NE, dout, mue, rse);
        bn_apply_kernel<<<gsz(nh2), 256, 0, stream>>>(hEmb, AX,    muh, rsh, bnhg, bnhb, hCurW, nh2, dout);
        bn_apply_kernel<<<gsz(ne2), 256, 0, stream>>>(eEmb, eGate, mue, rse, bneg, bneb, eCur,  ne2, dout);

        // 8) readout slice for this layer
        readout_kernel<<<gsz((size_t)B * dout), 256, 0, stream>>>(hCurW, dparent, dchild,
                                                                  dout, colOff, FT, B);
        Hcur = hCurW;
        din = dout;
        colOff += dout;
    }

    // ---- MLP head ----
    cat_cvt_kernel<<<gsz((size_t)B * 2 * FT), 256, 0, stream>>>(dparent, dchild, catBf, B, FT);
    transpose_cvt_bf16<<<gsz((size_t)(2 * FT) * FT), 256, 0, stream>>>(
        (const float*)d_in[76], Wt, 2 * FT, FT);
    launch_gemm(catBf, Wt, (const float*)d_in[77], dcenter, B, FT, 2 * FT, 1, stream);
    final_y_kernel<<<B, 256, 0, stream>>>(dcenter, (const float*)d_in[78],
                                          (const float*)d_in[79], dy, FT);
}
